// Prismatic_64845416235250
// MI455X (gfx1250) — compile-verified, matrix operand order checked by reading, not run on hardware
//
#include <hip/hip_runtime.h>

#define B_ 32
#define S_ 2048
#define H_ 1024
#define E_ 2
#define EPS_ 1e-5f
#define BAL_ 0.01f

#define SCHUNKS 32
#define SPER (S_ / SCHUNKS)

#define BM 128
#define BN 256
#define BK 32
#define NK (H_ / BK)
#define BKP 40   // padded LDS row stride in bf16 elements (80B: 16B-aligned, low-conflict)

typedef __attribute__((ext_vector_type(16))) __bf16 v16bf;
typedef __attribute__((ext_vector_type(8)))  float  v8f;

// one-instruction packed conversion: D[15:0]=bf16(lo), D[31:16]=bf16(hi)
__device__ __forceinline__ unsigned int pack2bf(float lo, float hi) {
    unsigned int d;
    asm("v_cvt_pk_bf16_f32 %0, %1, %2" : "=v"(d) : "v"(lo), "v"(hi));
    return d;
}

__device__ __forceinline__ unsigned short f2bf16(float f) {
    __bf16 a = (__bf16)f;
    return __builtin_bit_cast(unsigned short, a);
}

// CDNA5 async DMA: global -> LDS, 16B per active lane, tracked by ASYNCcnt
__device__ __forceinline__ void async_load_b128(unsigned lds_addr, unsigned gl_off,
                                                const void* base) {
    asm volatile("global_load_async_to_lds_b128 %0, %1, %2"
                 :
                 : "v"(lds_addr), "v"(gl_off), "s"(base)
                 : "memory");
}
__device__ __forceinline__ void wait_asynccnt0() {
    asm volatile("s_wait_asynccnt 0" ::: "memory");
}

// ---------------- Weight pre-pass: fp32 [e][k][n] -> bf16 transposed [e][n][k] ----------------
__global__ __launch_bounds__(256)
void moe_convert_wt(const float* __restrict__ ew, unsigned short* __restrict__ wt) {
    __shared__ float t[32][33];
    const int e  = blockIdx.z;
    const int k0 = blockIdx.x * 32;
    const int n0 = blockIdx.y * 32;
    const int tx = threadIdx.x & 31;
    const int ty = threadIdx.x >> 5;   // 0..7
    const float* src = ew + (size_t)e * H_ * H_;
    #pragma unroll
    for (int j = 0; j < 4; ++j) {
        const int k = ty * 4 + j;
        t[k][tx] = src[(size_t)(k0 + k) * H_ + n0 + tx];
    }
    __syncthreads();
    unsigned short* dst = wt + (size_t)e * H_ * H_;
    #pragma unroll
    for (int j = 0; j < 4; ++j) {
        const int n = ty * 4 + j;
        dst[(size_t)(n0 + n) * H_ + k0 + tx] = f2bf16(t[tx][n]);
    }
}

// ---------------- Router pass 1: partial sums over S chunks ----------------
__global__ void moe_router_partial(const float* __restrict__ x,
                                   float* __restrict__ partials) {
    const int b = blockIdx.x;
    const int c = blockIdx.y;
    const int h = threadIdx.x;                     // blockDim.x == H_ == 1024
    const float* p = x + ((size_t)b * S_ + (size_t)c * SPER) * H_ + h;
    float s = 0.f;
    #pragma unroll 4
    for (int i = 0; i < SPER; ++i) s += p[(size_t)i * H_];
    partials[((size_t)b * SCHUNKS + c) * H_ + h] = s;
}

// ---------------- block reduction (1024 threads, wave32) ----------------
__device__ __forceinline__ float block_reduce_sum(float v, float* red, int tid) {
    #pragma unroll
    for (int o = 16; o > 0; o >>= 1) v += __shfl_down(v, o, 32);
    const int wave = tid >> 5;
    if ((tid & 31) == 0) red[wave] = v;
    __syncthreads();
    float r = 0.f;
    if (wave == 0) {
        r = red[tid & 31];
        #pragma unroll
        for (int o = 16; o > 0; o >>= 1) r += __shfl_down(r, o, 32);
        if (tid == 0) red[0] = r;
    }
    __syncthreads();
    r = red[0];
    __syncthreads();
    return r;
}

// ---------------- Router pass 2: LN -> linear -> softmax -> argmax ----------------
__global__ void moe_router_finalize(const float* __restrict__ partials,
                                    const float* __restrict__ gamma,
                                    const float* __restrict__ beta,
                                    const float* __restrict__ rw,
                                    const float* __restrict__ rb,
                                    float* __restrict__ probs,
                                    int* __restrict__ eidx) {
    __shared__ float red[32];
    const int b = blockIdx.x;
    const int h = threadIdx.x;                     // blockDim.x == 1024

    float s = 0.f;
    #pragma unroll 4
    for (int c = 0; c < SCHUNKS; ++c)
        s += partials[((size_t)b * SCHUNKS + c) * H_ + h];
    const float x = s * (1.0f / (float)S_);        // seq_repr[b,h]

    const float mean = block_reduce_sum(x, red, h) * (1.0f / (float)H_);
    const float d    = x - mean;
    const float var  = block_reduce_sum(d * d, red, h) * (1.0f / (float)H_);
    const float nrm  = d * rsqrtf(var + EPS_) * gamma[h] + beta[h];

    const float l0 = block_reduce_sum(nrm * rw[h * E_ + 0], red, h);
    const float l1 = block_reduce_sum(nrm * rw[h * E_ + 1], red, h);

    if (h == 0) {
        const float z0 = l0 + rb[0];
        const float z1 = l1 + rb[1];
        const float m  = fmaxf(z0, z1);
        const float e0 = __expf(z0 - m);
        const float e1 = __expf(z1 - m);
        const float inv = 1.0f / (e0 + e1);
        const float p0 = e0 * inv, p1 = e1 * inv;
        probs[b * 2 + 0] = p0;
        probs[b * 2 + 1] = p1;
        eidx[b] = (p0 >= p1) ? 0 : 1;              // first max, like jnp.argmax
    }
}

// ---------------- aux balance loss ----------------
__global__ void moe_aux(const float* __restrict__ probs, float* __restrict__ out_aux) {
    if (threadIdx.x == 0) {
        float a0 = 0.f, a1 = 0.f;
        #pragma unroll 8
        for (int b = 0; b < B_; ++b) { a0 += probs[b * 2]; a1 += probs[b * 2 + 1]; }
        a0 *= (1.0f / (float)B_);
        a1 *= (1.0f / (float)B_);
        const float d0 = a0 - 0.5f, d1 = a1 - 0.5f;
        out_aux[0] = BAL_ * 0.5f * (d0 * d0 + d1 * d1);
    }
}

// ---------------- Expert batched GEMM: out[b] = X[b] @ W[idx[b]] + b[idx[b]] ----------------
// 128x256 block tile, 8 waves as 2x4, 64x64 per wave (4x4 WMMA frags).
// A: fp32 global -> regs -> v_cvt_pk_bf16_f32 -> LDS (double buffered).
// B: pre-transposed bf16 weights -> LDS via global_load_async_to_lds_b128 (ASYNCcnt).
__global__ __launch_bounds__(256)
void moe_expert_gemm(const float* __restrict__ x,
                     const unsigned short* __restrict__ wt,   // bf16 [e][n][k]
                     const float* __restrict__ eb,
                     const int* __restrict__ eidx,
                     float* __restrict__ out) {
    __shared__ __align__(16) unsigned short As[2][BM * BKP];   // [m][k] bf16
    __shared__ __align__(16) unsigned short Bs[2][BN * BKP];   // [n][k] bf16

    const int b    = blockIdx.z;
    const int m0   = blockIdx.x * BM;
    const int n0   = blockIdx.y * BN;
    const int tid  = threadIdx.x;
    const int lane = tid & 31;
    const int wave = tid >> 5;
    const int wm   = wave >> 2;     // 0..1 -> 64-row slab
    const int wn   = wave & 3;      // 0..3 -> 64-col slab

    const int idx = eidx[b];
    const float* gA = x + (size_t)b * S_ * H_;
    const unsigned short* gWt = wt + (size_t)idx * H_ * H_;

    v8f acc[4][4];
    #pragma unroll
    for (int i = 0; i < 4; ++i)
        #pragma unroll
        for (int j = 0; j < 4; ++j)
            acc[i][j] = (v8f){0.f, 0.f, 0.f, 0.f, 0.f, 0.f, 0.f, 0.f};

    const int rsel = lane & 15;
    const int koff = (lane < 16) ? 0 : 8;   // 16-bit fragment lane layout

    float4 stA[4];   // A staging: 4 float4 (128x32 fp32 / 256 threads)

    // A mapping: slot = tid + i*256 in [0,1024): row = slot/8, c4 = (slot%8)*4
    #define LOAD_A(K0)                                                                \
        {                                                                             \
            const int k0_ = (K0);                                                     \
            _Pragma("unroll")                                                         \
            for (int i = 0; i < 4; ++i) {                                             \
                const int slot = tid + i * 256;                                       \
                const int row  = slot >> 3;                                           \
                const int c4   = (slot & 7) << 2;                                     \
                stA[i] = *(const float4*)(gA + (size_t)(m0 + row) * H_ + k0_ + c4);   \
            }                                                                         \
        }

    #define STORE_A(BUF)                                                              \
        {                                                                             \
            const int buf_ = (BUF);                                                   \
            _Pragma("unroll")                                                         \
            for (int i = 0; i < 4; ++i) {                                             \
                const int slot = tid + i * 256;                                       \
                const int row  = slot >> 3;                                           \
                const int c4   = (slot & 7) << 2;                                     \
                uint2 p;                                                              \
                p.x = pack2bf(stA[i].x, stA[i].y);                                    \
                p.y = pack2bf(stA[i].z, stA[i].w);                                    \
                *(uint2*)&As[buf_][row * BKP + c4] = p;                               \
            }                                                                         \
        }

    // B mapping (async, 16B per lane): chunk = tid + i*256 in [0,1024):
    // n = chunk/4, kq = (chunk%4)*8 elements
    #define ASYNC_B(BUF, K0)                                                          \
        {                                                                             \
            const int buf_ = (BUF);                                                   \
            const int k0_  = (K0);                                                    \
            _Pragma("unroll")                                                         \
            for (int i = 0; i < 4; ++i) {                                             \
                const int chunk = tid + i * 256;                                      \
                const int n  = chunk >> 2;                                            \
                const int kq = (chunk & 3) << 3;                                      \
                const unsigned lds  = (unsigned)(size_t)&Bs[buf_][n * BKP + kq];      \
                const unsigned goff = (unsigned)((((n0 + n) * H_) + k0_ + kq) * 2);   \
                async_load_b128(lds, goff, gWt);                                      \
            }                                                                         \
        }

    // prologue: fill buffer 0
    ASYNC_B(0, 0)
    LOAD_A(0)
    STORE_A(0)
    wait_asynccnt0();
    __syncthreads();

    for (int kt = 0; kt < NK; ++kt) {
        const int cur = kt & 1;

        // issue next tile's transfers early (latency hidden behind 16 WMMAs)
        if (kt + 1 < NK) {
            ASYNC_B(cur ^ 1, (kt + 1) * BK)
            LOAD_A((kt + 1) * BK)
            if (kt + 2 < NK && tid < 128)   // keep L2 warm two tiles ahead (A only)
                __builtin_prefetch(gA + (size_t)(m0 + tid) * H_ + (kt + 2) * BK, 0, 0);
        }

        // ---- fragments from LDS: two aligned b128 loads each ----
        union Frag { unsigned int u[8]; v16bf v; };
        Frag afr[4], bfr[4];
        #pragma unroll
        for (int ni = 0; ni < 4; ++ni) {
            const unsigned short* src = &Bs[cur][(wn * 64 + ni * 16 + rsel) * BKP + koff];
            *(uint4*)&bfr[ni].u[0] = *(const uint4*)(src);        // K 0..7 / 8..15
            *(uint4*)&bfr[ni].u[4] = *(const uint4*)(src + 16);   // K 16..23 / 24..31
        }
        #pragma unroll
        for (int mi = 0; mi < 4; ++mi) {
            const unsigned short* src = &As[cur][(wm * 64 + mi * 16 + rsel) * BKP + koff];
            *(uint4*)&afr[mi].u[0] = *(const uint4*)(src);
            *(uint4*)&afr[mi].u[4] = *(const uint4*)(src + 16);
        }

        #pragma unroll
        for (int mi = 0; mi < 4; ++mi)
            #pragma unroll
            for (int ni = 0; ni < 4; ++ni)
                acc[mi][ni] = __builtin_amdgcn_wmma_f32_16x16x32_bf16(
                    false, afr[mi].v, false, bfr[ni].v,
                    (short)0, acc[mi][ni], false, false);

        // convert + stage next A tile into the other buffer (overlaps XDL execution)
        if (kt + 1 < NK) STORE_A(cur ^ 1)

        wait_asynccnt0();      // B tile for next iter complete (per-wave)
        __syncthreads();       // all waves' async writes + A stores visible
    }

    // ---- epilogue: + bias, store fp32 ----
    const float* bias = eb + (size_t)idx * H_;
    float* gO = out + (size_t)b * S_ * H_;
    #pragma unroll
    for (int mi = 0; mi < 4; ++mi) {
        const int mbase = m0 + wm * 64 + mi * 16 + ((lane >> 4) << 3);
        #pragma unroll
        for (int ni = 0; ni < 4; ++ni) {
            const int gn = n0 + wn * 64 + ni * 16 + (lane & 15);
            const float bv = bias[gn];
            #pragma unroll
            for (int r = 0; r < 8; ++r)
                gO[(size_t)(mbase + r) * H_ + gn] = acc[mi][ni][r] + bv;
        }
    }
    #undef LOAD_A
    #undef STORE_A
    #undef ASYNC_B
}

extern "C" void kernel_launch(void* const* d_in, const int* in_sizes, int n_in,
                              void* d_out, int out_size, void* d_ws, size_t ws_size,
                              hipStream_t stream) {
    const float* x  = (const float*)d_in[0];   // [B,S,H]
    const float* g  = (const float*)d_in[1];   // [H]
    const float* be = (const float*)d_in[2];   // [H]
    const float* rw = (const float*)d_in[3];   // [H,E]
    const float* rb = (const float*)d_in[4];   // [E]
    const float* ew = (const float*)d_in[5];   // [E,H,H]
    const float* eb = (const float*)d_in[6];   // [E,H]
    float* out = (float*)d_out;                // [B,S,H] + 1 aux

    char* ws = (char*)d_ws;
    const size_t partials_bytes = (size_t)B_ * SCHUNKS * H_ * 4;        // 4 MB
    float*          partials = (float*)ws;
    unsigned short* wt       = (unsigned short*)(ws + partials_bytes);  // 4 MB bf16 [e][n][k]
    float*          probs    = (float*)(ws + partials_bytes + (size_t)E_ * H_ * H_ * 2);
    int*            eidx     = (int*)(probs + B_ * E_);

    moe_convert_wt<<<dim3(H_ / 32, H_ / 32, E_), 256, 0, stream>>>(ew, wt);
    moe_router_partial<<<dim3(B_, SCHUNKS), 1024, 0, stream>>>(x, partials);
    moe_router_finalize<<<dim3(B_), 1024, 0, stream>>>(partials, g, be, rw, rb, probs, eidx);
    moe_aux<<<1, 32, 0, stream>>>(probs, out + (size_t)B_ * S_ * H_);
    moe_expert_gemm<<<dim3(S_ / BM, H_ / BN, B_), 256, 0, stream>>>(x, wt, eb, eidx, out);
}
